// HE2RNA_40621800685696
// MI455X (gfx1250) — matrix operand order repaired and checked by reading, not compiled, not run on hardware
//
#include <hip/hip_runtime.h>
#include <hip/hip_bf16.h>

typedef __attribute__((ext_vector_type(16))) _Float16 v16h;
typedef __attribute__((ext_vector_type(8)))  float    v8f;

#define CLAMP_V 1.0e4f

// ---- workspace layout (bytes), all offsets 256-aligned ----
static constexpr size_t O_W0   = 0;                          // 256*2048 f16 (1 MB)
static constexpr size_t O_W1   = O_W0 + (size_t)256*2048*2;  // 256*256 f16
static constexpr size_t O_W2   = O_W1 + (size_t)256*256*2;   // 1008*256 f16 (padded)
static constexpr size_t O_MASK = O_W2 + (size_t)1008*256*2;  // 8*8000 f32
static constexpr size_t O_Y    = O_MASK + (size_t)8*8000*4;  // 8*1000*8000 f32

// -------------------- weight conversion --------------------
__global__ void cvt_weights(const float* __restrict__ W0, const float* __restrict__ W1,
                            const float* __restrict__ W2,
                            _Float16* __restrict__ w0, _Float16* __restrict__ w1,
                            _Float16* __restrict__ w2)
{
  int i = blockIdx.x * 256 + threadIdx.x;
  int stride = gridDim.x * 256;
  for (int j = i; j < 256 * 2048; j += stride) w0[j] = (_Float16)W0[j];
  for (int j = i; j < 256 * 256;  j += stride) w1[j] = (_Float16)W1[j];
  for (int j = i; j < 1008 * 256; j += stride)
    w2[j] = (j < 1000 * 256) ? (_Float16)W2[j] : (_Float16)0.0f;
}

// -------------------- WMMA fragment loaders --------------------
// A (16x32 f16): lanes 0-15 -> M=lane, K in {kb..kb+7, kb+16..kb+23}, kb = (lane>>4)*8
__device__ __forceinline__ v16h ldA(const _Float16* W, int ldk, int row0, int k0, int lane)
{
  int m  = lane & 15;
  int kb = (lane >> 4) << 3;
  const _Float16* p = W + (size_t)(row0 + m) * ldk + k0 + kb;
  v16h a;
  ((uint4*)&a)[0] = *(const uint4*)(p);
  ((uint4*)&a)[1] = *(const uint4*)(p + 16);
  return a;
}
// B (32x16 f16) from LDS stored [n][k]: lane holds column n=lane&15, K in {kb..kb+7, kb+16..kb+23}, kb = (lane>>4)*8
__device__ __forceinline__ v16h ldB(const _Float16* L, int ld, int col0, int k0, int lane)
{
  const _Float16* p = L + (size_t)(col0 + (lane & 15)) * ld + k0 + ((lane >> 4) << 3);
  v16h b;
  ((uint4*)&b)[0] = *(const uint4*)(p);
  ((uint4*)&b)[1] = *(const uint4*)(p + 16);
  return b;
}
#define WMMA(acc, a, b) \
  acc = __builtin_amdgcn_wmma_f32_16x16x32_f16(false, a, false, b, (short)0, acc, false, false)

// -------------------- fused conv stack --------------------
// grid = 8 batches * 250 column tiles; block = 256 threads (8 waves)
__global__ __launch_bounds__(256, 2)
void he2rna_fused(const float* __restrict__ x,
                  const _Float16* __restrict__ w0, const float* __restrict__ b0,
                  const _Float16* __restrict__ w1, const float* __restrict__ b1,
                  const _Float16* __restrict__ w2, const float* __restrict__ b2,
                  float* __restrict__ maskg, float* __restrict__ y)
{
  __shared__ _Float16 ldsX[2][32 * 40];  // double-buffered x tile, [n][k] stride 40
  __shared__ _Float16 ldsH0[32 * 264];   // h0, [n][k] stride 264
  __shared__ _Float16 ldsH1[32 * 264];   // h1
  __shared__ float    red[32 * 32];      // column-max reduction
  __shared__ float    cmask[32];

  const int t    = threadIdx.x;
  const int lane = t & 31;
  const int w    = t >> 5;
  const int blk  = blockIdx.x;
  const int bb   = blk / 250;
  const int n0   = (blk % 250) * 32;

  // x staging decomposition: thread t loads float4 at (k = t>>3, n = (t&7)*4)
  const int n4 = (t & 7) * 4;
  const int kk = t >> 3;                 // 0..31
  float4 cm = make_float4(-1.f, -1.f, -1.f, -1.f);

  const int r0 = w * 32;                 // this wave's 32 output rows (GEMM0/GEMM1)
  v8f a00 = {}, a01 = {}, a10 = {}, a11 = {};

  const float* xb = x + ((size_t)bb * 2048) * 8000 + n0;

  // ---- GEMM0: h0 = W0 @ x, K = 2048 in 64 chunks of 32, double buffered ----
  float4 xv = *(const float4*)(xb + (size_t)kk * 8000 + n4);  // chunk 0
  for (int c = 0; c < 64; ++c) {
    const int k0 = c * 32;
    _Float16* bx = &ldsX[c & 1][0];

    cm.x = fmaxf(cm.x, xv.x); cm.y = fmaxf(cm.y, xv.y);
    cm.z = fmaxf(cm.z, xv.z); cm.w = fmaxf(cm.w, xv.w);
    bx[(n4 + 0) * 40 + kk] = (_Float16)xv.x;
    bx[(n4 + 1) * 40 + kk] = (_Float16)xv.y;
    bx[(n4 + 2) * 40 + kk] = (_Float16)xv.z;
    bx[(n4 + 3) * 40 + kk] = (_Float16)xv.w;
    __syncthreads();

    float4 xvn;
    if (c < 63) {
      xvn = *(const float4*)(xb + (size_t)(k0 + 32 + kk) * 8000 + n4);
      if (c < 62)
        __builtin_prefetch(xb + (size_t)(k0 + 64 + kk) * 8000 + n4, 0, 3);
    } else {
      xvn = xv;
    }

    v16h af0 = ldA(w0, 2048, r0,      k0, lane);
    v16h af1 = ldA(w0, 2048, r0 + 16, k0, lane);
    v16h bf0 = ldB(bx, 40, 0,  0, lane);
    v16h bf1 = ldB(bx, 40, 16, 0, lane);
    WMMA(a00, af0, bf0);
    WMMA(a01, af0, bf1);
    WMMA(a10, af1, bf0);
    WMMA(a11, af1, bf1);
    xv = xvn;
  }

  // ---- per-column mask = (max_c x > 0) ----
  red[kk * 32 + n4 + 0] = cm.x;
  red[kk * 32 + n4 + 1] = cm.y;
  red[kk * 32 + n4 + 2] = cm.z;
  red[kk * 32 + n4 + 3] = cm.w;
  __syncthreads();
  if (t < 32) {
    float m = red[t];
    #pragma unroll
    for (int j = 1; j < 32; ++j) m = fmaxf(m, red[j * 32 + t]);
    float mk = (m > 0.0f) ? 1.0f : 0.0f;
    cmask[t] = mk;
    maskg[(size_t)bb * 8000 + n0 + t] = mk;
  }

  // ---- bias + relu + clamp -> ldsH0 ----
  {
    int mrow = (lane >> 4) * 8;
    int ncl  = lane & 15;
    #pragma unroll
    for (int r = 0; r < 8; ++r) {
      int m0_ = r0 + mrow + r;
      int m1_ = r0 + 16 + mrow + r;
      float h;
      h = fminf(fmaxf(a00[r] + b0[m0_], 0.f), CLAMP_V); ldsH0[ncl * 264 + m0_]        = (_Float16)h;
      h = fminf(fmaxf(a01[r] + b0[m0_], 0.f), CLAMP_V); ldsH0[(16 + ncl) * 264 + m0_] = (_Float16)h;
      h = fminf(fmaxf(a10[r] + b0[m1_], 0.f), CLAMP_V); ldsH0[ncl * 264 + m1_]        = (_Float16)h;
      h = fminf(fmaxf(a11[r] + b0[m1_], 0.f), CLAMP_V); ldsH0[(16 + ncl) * 264 + m1_] = (_Float16)h;
    }
  }
  __syncthreads();

  // ---- GEMM1: h1 = W1 @ h0, K = 256 ----
  v8f c00 = {}, c01 = {}, c10 = {}, c11 = {};
  #pragma unroll
  for (int k0 = 0; k0 < 256; k0 += 32) {
    v16h af0 = ldA(w1, 256, r0,      k0, lane);
    v16h af1 = ldA(w1, 256, r0 + 16, k0, lane);
    v16h bf0 = ldB(ldsH0, 264, 0,  k0, lane);
    v16h bf1 = ldB(ldsH0, 264, 16, k0, lane);
    WMMA(c00, af0, bf0);
    WMMA(c01, af0, bf1);
    WMMA(c10, af1, bf0);
    WMMA(c11, af1, bf1);
  }
  {
    int mrow = (lane >> 4) * 8;
    int ncl  = lane & 15;
    #pragma unroll
    for (int r = 0; r < 8; ++r) {
      int m0_ = r0 + mrow + r;
      int m1_ = r0 + 16 + mrow + r;
      float h;
      h = fminf(fmaxf(c00[r] + b1[m0_], 0.f), CLAMP_V); ldsH1[ncl * 264 + m0_]        = (_Float16)h;
      h = fminf(fmaxf(c01[r] + b1[m0_], 0.f), CLAMP_V); ldsH1[(16 + ncl) * 264 + m0_] = (_Float16)h;
      h = fminf(fmaxf(c10[r] + b1[m1_], 0.f), CLAMP_V); ldsH1[ncl * 264 + m1_]        = (_Float16)h;
      h = fminf(fmaxf(c11[r] + b1[m1_], 0.f), CLAMP_V); ldsH1[(16 + ncl) * 264 + m1_] = (_Float16)h;
    }
  }
  __syncthreads();

  // ---- GEMM2: y = (W2 @ h1 + b2) * mask, O padded to 1008 (63 row tiles) ----
  for (int tr = w; tr < 63; tr += 8) {
    v8f d0 = {}, d1 = {};
    #pragma unroll
    for (int k0 = 0; k0 < 256; k0 += 32) {
      v16h af  = ldA(w2, 256, tr * 16, k0, lane);
      v16h bf0 = ldB(ldsH1, 264, 0,  k0, lane);
      v16h bf1 = ldB(ldsH1, 264, 16, k0, lane);
      WMMA(d0, af, bf0);
      WMMA(d1, af, bf1);
    }
    int mrow = (lane >> 4) * 8;
    int ncl  = lane & 15;
    #pragma unroll
    for (int r = 0; r < 8; ++r) {
      int o = tr * 16 + mrow + r;
      if (o < 1000) {
        float bv = b2[o];
        size_t base = ((size_t)bb * 1000 + o) * 8000 + n0;
        y[base + ncl]      = (d0[r] + bv) * cmask[ncl];
        y[base + 16 + ncl] = (d1[r] + bv) * cmask[16 + ncl];
      }
    }
  }
}

// -------------------- top-k + prediction --------------------
__device__ __forceinline__ unsigned orderKey(float f)
{
  unsigned b = __float_as_uint(f);
  return (b & 0x80000000u) ? ~b : (b | 0x80000000u);
}
__device__ __forceinline__ float invKey(unsigned k)
{
  unsigned b = (k & 0x80000000u) ? (k & 0x7FFFFFFFu) : ~k;
  return __uint_as_float(b);
}

// one block per (b, o) row of y
__global__ __launch_bounds__(256, 2)
void topk_pred(const float* __restrict__ y, const float* __restrict__ mask,
               float* __restrict__ out)
{
  __shared__ unsigned keys[8192];
  __shared__ int      hist[256];
  __shared__ unsigned selk[128];
  __shared__ int      scnt;
  __shared__ unsigned s_prefix;
  __shared__ int      s_remaining;

  const int t   = threadIdx.x;
  const int row = blockIdx.x;        // b*1000 + o
  const int b   = row / 1000;

  const float* yr = y + (size_t)row * 8000;
  for (int i = t; i < 8192; i += 256)
    keys[i] = (i < 8000) ? orderKey(yr[i]) : 0u;
  if (t == 0) { s_prefix = 0u; s_remaining = 100; }
  if (t < 128) selk[t] = 0u;
  __syncthreads();

  // 4-pass radix select: find key of the 100th-largest element
  unsigned prefmask = 0u;
  for (int shift = 24; shift >= 0; shift -= 8) {
    hist[t] = 0;
    __syncthreads();
    unsigned pref = s_prefix;
    for (int i = t; i < 8192; i += 256) {
      unsigned k = keys[i];
      if ((k & prefmask) == pref)
        atomicAdd(&hist[(k >> shift) & 255], 1);
    }
    __syncthreads();
    if (t == 0) {
      int rem = s_remaining;
      int bin = 255;
      for (; bin > 0; --bin) {
        int c = hist[bin];
        if (c >= rem) break;
        rem -= c;
      }
      s_remaining = rem;
      s_prefix = pref | ((unsigned)bin << shift);
    }
    prefmask |= (0xFFu << shift);
    __syncthreads();
  }

  const unsigned T = s_prefix;       // exact 100th-largest key
  if (t == 0) scnt = 0;
  __syncthreads();
  for (int i = t; i < 8192; i += 256) {
    unsigned k = keys[i];
    if (k >= T) {
      int p = atomicAdd(&scnt, 1);
      if (p < 128) selk[p] = k;
    }
  }
  __syncthreads();

  // bitonic sort 128 keys, descending
  for (int ksz = 2; ksz <= 128; ksz <<= 1) {
    for (int j = ksz >> 1; j > 0; j >>= 1) {
      if (t < 128) {
        int ixj = t ^ j;
        if (ixj > t) {
          unsigned a = selk[t], c = selk[ixj];
          bool down = ((t & ksz) == 0);
          if (down ? (a < c) : (a > c)) { selk[t] = c; selk[ixj] = a; }
        }
      }
      __syncthreads();
    }
  }

  if (t == 0) {
    const float* mrow = mask + (size_t)b * 8000;   // positional mask[:, :, :k]
    float num = 0.f, den = 0.f, pred = 0.f;
    for (int i = 0; i < 100; ++i) {
      float v = invKey(selk[i]);
      float m = mrow[i];
      num += v * m;
      den += m;
      if (i == 9 || i == 24 || i == 49 || i == 99)
        pred += (num / den) * 0.25f;
    }
    out[row] = pred;
  }
}

// -------------------- launch --------------------
extern "C" void kernel_launch(void* const* d_in, const int* in_sizes, int n_in,
                              void* d_out, int out_size, void* d_ws, size_t ws_size,
                              hipStream_t stream)
{
  const float* x  = (const float*)d_in[0];
  const float* W0 = (const float*)d_in[1];
  const float* b0 = (const float*)d_in[2];
  const float* W1 = (const float*)d_in[3];
  const float* b1 = (const float*)d_in[4];
  const float* W2 = (const float*)d_in[5];
  const float* b2 = (const float*)d_in[6];

  char* ws = (char*)d_ws;
  _Float16* w0h = (_Float16*)(ws + O_W0);
  _Float16* w1h = (_Float16*)(ws + O_W1);
  _Float16* w2h = (_Float16*)(ws + O_W2);
  float* msk = (float*)(ws + O_MASK);
  float* yb  = (float*)(ws + O_Y);

  cvt_weights<<<512, 256, 0, stream>>>(W0, W1, W2, w0h, w1h, w2h);
  he2rna_fused<<<2000, 256, 0, stream>>>(x, w0h, b0, w1h, b1, w2h, b2, msk, yb);
  topk_pred<<<8000, 256, 0, stream>>>(yb, msk, (float*)d_out);
}
